// EGNNBlock_63651415326947
// MI455X (gfx1250) — matrix operand order repaired and checked by reading, not compiled, not run on hardware
//
#include <hip/hip_runtime.h>

#define N_NODES 50000
#define N_EDGES 600000
#define HDIM    128
#define EDIM    16
#define TDIM    32
#define KPAD    320
#define NW      8      // waves per block (wave32)

typedef __attribute__((ext_vector_type(16))) __bf16 v16bf;
typedef __attribute__((ext_vector_type(8)))  __bf16 v8bf;
typedef __attribute__((ext_vector_type(8)))  float  v8f;
typedef int v4i __attribute__((vector_size(16)));

#define AS_G __attribute__((address_space(1)))
#define AS_L __attribute__((address_space(3)))

#if defined(__has_builtin)
#if __has_builtin(__builtin_amdgcn_global_load_async_to_lds_b128) && \
    __has_builtin(__builtin_amdgcn_s_wait_asynccnt)
#define USE_ASYNC_LDS 1
#endif
#endif

__device__ __forceinline__ __bf16 f2bf(float f) {
  unsigned u = __builtin_bit_cast(unsigned, f);
  u += 0x7FFFu + ((u >> 16) & 1u);              // round-to-nearest-even
  unsigned short s = (unsigned short)(u >> 16);
  return __builtin_bit_cast(__bf16, s);
}
__device__ __forceinline__ float siluf(float x) { return x / (1.f + __expf(-x)); }
__device__ __forceinline__ float sigmf(float x) { return 1.f / (1.f + __expf(-x)); }

__device__ __forceinline__ void atomAddF(float* p, float v) {
  __hip_atomic_fetch_add(p, v, __ATOMIC_RELAXED, __HIP_MEMORY_SCOPE_AGENT);
}

// 16-byte global->LDS copy: async (ASYNCcnt-tracked) on CDNA5 when the builtin exists.
__device__ __forceinline__ void cp16(const __bf16* g, __bf16* l) {
#ifdef USE_ASYNC_LDS
  __builtin_amdgcn_global_load_async_to_lds_b128((AS_G v4i*)g, (AS_L v4i*)l, 0, 0);
#else
  *(uint4*)l = *(const uint4*)g;
#endif
}
__device__ __forceinline__ void cp_fence() {
#ifdef USE_ASYNC_LDS
  __builtin_amdgcn_s_wait_asynccnt(0);
#endif
}

__device__ __forceinline__ v16bf cat16(v8bf lo, v8bf hi) {
  return __builtin_shufflevector(lo, hi, 0,1,2,3,4,5,6,7,8,9,10,11,12,13,14,15);
}

// A fragment: 16x32 bf16 tile from row-major LDS matrix A (row stride rs elems).
// lanes 0-15: row m=lane, K {k0..k0+7, k0+16..k0+23}; lanes 16-31: K {+8, +24}.
__device__ __forceinline__ v16bf load_a_frag(const __bf16* A, int lane, int k0, int rs) {
  int m = lane & 15, g = (lane >> 4) & 1;
  const __bf16* p = A + m * rs + k0 + g * 8;
  return cat16(*(const v8bf*)p, *(const v8bf*)(p + 16));
}

// B fragment: packed tile, lane l holds 16 contiguous bf16 at tile + l*16.
__device__ __forceinline__ v16bf load_b_frag(const __bf16* tile, int lane) {
  const __bf16* p = tile + lane * 16;
  return cat16(*(const v8bf*)p, *(const v8bf*)(p + 8));
}

__device__ __forceinline__ v8f wmma_bf16(v16bf a, v16bf b, v8f c) {
  return __builtin_amdgcn_wmma_f32_16x16x32_bf16(false, a, false, b, (short)0, c, false, false);
}

__device__ __forceinline__ void st4bf(__bf16* p, float4 v) {
  p[0] = f2bf(v.x); p[1] = f2bf(v.y); p[2] = f2bf(v.z); p[3] = f2bf(v.w);
}

__device__ __forceinline__ void copyW(__bf16* dst, const __bf16* src, int n, int tid, int nt) {
  const uint4* s = (const uint4*)src;
  uint4* d = (uint4*)dst;
  for (int i = tid; i < (n >> 3); i += nt) d[i] = s[i];
}
__device__ __forceinline__ void copyF(float* dst, const float* src, int n, int tid, int nt) {
  for (int i = tid; i < n; i += nt) dst[i] = src[i];
}

// ---------------- init / preprocessing ----------------
__global__ void k_zero(float* __restrict__ p, int n) {
  int i = blockIdx.x * blockDim.x + threadIdx.x;
  if (i < n) p[i] = 0.f;
}

__global__ void k_cvt(const float* __restrict__ s, __bf16* __restrict__ d, int n) {
  int i = blockIdx.x * blockDim.x + threadIdx.x;
  if (i < n) d[i] = f2bf(s[i]);
}

// ---------------- weight repack: f32 row-major (Ksrc x 128) -> bf16 WMMA-B tiles ----------------
// mode 0: identity; mode 1: We_w1 (pad-320, dist_sq src row 256 -> padded 304);
// mode 2: Wx_w1 (pad-320, rows >=304 zero)
__global__ void k_repack(const float* __restrict__ src, __bf16* __restrict__ dst,
                         int Kpad, int Ksrc, int mode) {
  int e = blockIdx.x * blockDim.x + threadIdx.x;
  int total = Kpad * HDIM;
  if (e >= total) return;
  int tile = e >> 9, w = e & 511;
  int lane = w >> 4, j = w & 15;
  int kc = tile >> 3, nt = tile & 7;
  int kp = kc * 32 + ((lane >> 4) & 1) * 8 + (j < 8 ? j : 8 + j);
  int n  = nt * 16 + (lane & 15);
  int ks;
  if (mode == 1)      ks = (kp < 256) ? kp : (kp < 304) ? kp + 1 : (kp == 304) ? 256 : -1;
  else if (mode == 2) ks = (kp < 304) ? kp : -1;
  else                ks = kp;
  float v = (ks >= 0 && ks < Ksrc) ? src[(long)ks * HDIM + n] : 0.f;
  dst[e] = f2bf(v);
}

// Build the shared 16x320 bf16 A tile for one wave (both edge kernels).
__device__ __forceinline__ void build_edge_A(__bf16* A, const __bf16* hbf,
                                             const __bf16* eabf, const __bf16* tbf,
                                             const int* src16, const int* dst16,
                                             long e0, int lane) {
  // h_src / h_dst: 16 rows x 16 chunks of 16B each
  for (int i = lane; i < 256; i += 32) {
    int m = i >> 4, c = i & 15;
    cp16(hbf + (size_t)src16[m] * HDIM + c * 8, A + m * KPAD + c * 8);
    cp16(hbf + (size_t)dst16[m] * HDIM + c * 8, A + m * KPAD + HDIM + c * 8);
  }
  // edge_attr -> cols 256..271 (16 rows x 2 chunks = 32 = one per lane)
  {
    int m = lane >> 1, c = lane & 1;
    cp16(eabf + (e0 + m) * EDIM + c * 8, A + m * KPAD + 256 + c * 8);
  }
  // t_emb[src] -> cols 272..303 (16 rows x 4 chunks)
  for (int i = lane; i < 64; i += 32) {
    int m = i >> 2, c = i & 3;
    cp16(tbf + (size_t)src16[m] * TDIM + c * 8, A + m * KPAD + 272 + c * 8);
  }
  cp_fence();
}

// ---------------- edge kernel A: phi_e + attention + msg scatter ----------------
__global__ __launch_bounds__(NW * 32, 1)
void k_edge_msg(const float* __restrict__ x,
                const long long* __restrict__ eidx,
                const __bf16* __restrict__ hbf, const __bf16* __restrict__ eabf,
                const __bf16* __restrict__ tbf,
                const __bf16* __restrict__ W1f, const __bf16* __restrict__ W2f,
                const float* __restrict__ b1, const float* __restrict__ b2,
                const float* __restrict__ wAtt, const float* __restrict__ attB,
                float* __restrict__ msg_agg) {
  __shared__ alignas(16) __bf16 sW1[KPAD * HDIM];       // 80 KB
  __shared__ alignas(16) __bf16 sW2[HDIM * HDIM];       // 32 KB
  __shared__ float sB1[HDIM], sB2[HDIM], sWatt[HDIM];
  __shared__ alignas(16) __bf16 sA[NW][16 * KPAD];      // 80 KB
  __shared__ alignas(16) __bf16 sHid[NW][16 * HDIM];    // 32 KB
  __shared__ alignas(16) float  sM[NW][16 * HDIM];      // 64 KB
  __shared__ int   sSrc[NW][16], sDst[NW][16];
  __shared__ float sAtt[NW][16];

  const int tid = threadIdx.x, lane = tid & 31, w = tid >> 5;
  copyW(sW1, W1f, KPAD * HDIM, tid, NW * 32);
  copyW(sW2, W2f, HDIM * HDIM, tid, NW * 32);
  copyF(sB1, b1, HDIM, tid, NW * 32);
  copyF(sB2, b2, HDIM, tid, NW * 32);
  copyF(sWatt, wAtt, HDIM, tid, NW * 32);
  __syncthreads();
  const float attBias = attB[0];

  __bf16* A  = &sA[w][0];
  __bf16* Hd = &sHid[w][0];
  float*  Mb = &sM[w][0];

  // zero the K-pad columns (304..319) once; col 304 is rewritten per tile (dist_sq)
  for (int i = lane; i < 256; i += 32) {
    int m = i >> 4, k = 304 + (i & 15);
    A[m * KPAD + k] = f2bf(0.f);
  }

  const int nTiles = N_EDGES / 16;
  const int gw = blockIdx.x * NW + w;
  const int stride = gridDim.x * NW;

  for (int t = gw; t < nTiles; t += stride) {
    const long e0 = (long)t * 16;
    if (lane < 16) {
      long e = e0 + lane;
      sSrc[w][lane] = (int)eidx[e];
      sDst[w][lane] = (int)eidx[N_EDGES + e];
    }
    __builtin_prefetch(eabf + (e0 + (long)stride * 16) * EDIM, 0, 1);
    __builtin_prefetch(eidx + e0 + (long)stride * 16, 0, 1);

    build_edge_A(A, hbf, eabf, tbf, sSrc[w], sDst[w], e0, lane);

    // dist_sq -> col 304
    if (lane < 16) {
      int s = sSrc[w][lane], d = sDst[w][lane];
      float dx = x[s * 3 + 0] - x[d * 3 + 0];
      float dy = x[s * 3 + 1] - x[d * 3 + 1];
      float dz = x[s * 3 + 2] - x[d * 3 + 2];
      A[lane * KPAD + 304] = f2bf(dx * dx + dy * dy + dz * dz);
    }

    // layer 1: SiLU(A @ W1 + b1) -> Hd (bf16)
    for (int nt = 0; nt < 8; ++nt) {
      v8f acc = {};
      for (int kc = 0; kc < 10; ++kc) {
        v16bf a = load_a_frag(A, lane, kc * 32, KPAD);
        v16bf b = load_b_frag(sW1 + (kc * 8 + nt) * 512, lane);
        acc = wmma_bf16(a, b, acc);
      }
      int col = nt * 16 + (lane & 15);
      int mBase = (lane >> 4) * 8;
      float bb = sB1[col];
#pragma unroll
      for (int j = 0; j < 8; ++j)
        Hd[(mBase + j) * HDIM + col] = f2bf(siluf(acc[j] + bb));
    }
    // layer 2: m_ij = Hd @ W2 + b2 -> Mb (f32)
    for (int nt = 0; nt < 8; ++nt) {
      v8f acc = {};
      for (int kc = 0; kc < 4; ++kc) {
        v16bf a = load_a_frag(Hd, lane, kc * 32, HDIM);
        v16bf b = load_b_frag(sW2 + (kc * 8 + nt) * 512, lane);
        acc = wmma_bf16(a, b, acc);
      }
      int col = nt * 16 + (lane & 15);
      int mBase = (lane >> 4) * 8;
      float bb = sB2[col];
#pragma unroll
      for (int j = 0; j < 8; ++j)
        Mb[(mBase + j) * HDIM + col] = acc[j] + bb;
    }
    // attention gate per edge row
    if (lane < 16) {
      float dot = attBias;
      const float* row = Mb + lane * HDIM;
      for (int n = 0; n < HDIM; ++n) dot += row[n] * sWatt[n];
      sAtt[w][lane] = sigmf(dot);
    }
    // scatter att * m_ij into msg_agg[dst]
    for (int i = lane; i < 16 * HDIM; i += 32) {
      int m = i >> 7;
      float v = Mb[i] * sAtt[w][m];
      atomAddF(&msg_agg[(long)sDst[w][m] * HDIM + (i & 127)], v);
    }
  }
}

// ---------------- edge kernel B: phi_x + coord scatter ----------------
__global__ __launch_bounds__(NW * 32, 1)
void k_edge_coord(const float* __restrict__ x,
                  const long long* __restrict__ eidx,
                  const __bf16* __restrict__ hbf, const __bf16* __restrict__ eabf,
                  const __bf16* __restrict__ tbf,
                  const __bf16* __restrict__ W1f, const __bf16* __restrict__ W2f,
                  const float* __restrict__ b1, const float* __restrict__ b2,
                  const float* __restrict__ w3,
                  float* __restrict__ coord_agg) {
  __shared__ alignas(16) __bf16 sW1[KPAD * HDIM];
  __shared__ alignas(16) __bf16 sW2[HDIM * HDIM];
  __shared__ float sB1[HDIM], sB2[HDIM], sW3[HDIM];
  __shared__ alignas(16) __bf16 sA[NW][16 * KPAD];
  __shared__ alignas(16) __bf16 sHid[NW][16 * HDIM];
  __shared__ alignas(16) float  sM[NW][16 * HDIM];
  __shared__ int   sSrc[NW][16], sDst[NW][16];
  __shared__ float sCw[NW][16];
  __shared__ float sDiff[NW][16][4];

  const int tid = threadIdx.x, lane = tid & 31, w = tid >> 5;
  copyW(sW1, W1f, KPAD * HDIM, tid, NW * 32);
  copyW(sW2, W2f, HDIM * HDIM, tid, NW * 32);
  copyF(sB1, b1, HDIM, tid, NW * 32);
  copyF(sB2, b2, HDIM, tid, NW * 32);
  copyF(sW3, w3, HDIM, tid, NW * 32);
  __syncthreads();

  __bf16* A  = &sA[w][0];
  __bf16* Hd = &sHid[w][0];
  float*  Mb = &sM[w][0];

  // cols 304..319 are zero rows in Wx1f packing; keep A zero there (avoid NaN*0)
  for (int i = lane; i < 256; i += 32) {
    int m = i >> 4, k = 304 + (i & 15);
    A[m * KPAD + k] = f2bf(0.f);
  }

  const int nTiles = N_EDGES / 16;
  const int gw = blockIdx.x * NW + w;
  const int stride = gridDim.x * NW;

  for (int t = gw; t < nTiles; t += stride) {
    const long e0 = (long)t * 16;
    if (lane < 16) {
      long e = e0 + lane;
      sSrc[w][lane] = (int)eidx[e];
      sDst[w][lane] = (int)eidx[N_EDGES + e];
    }
    __builtin_prefetch(eabf + (e0 + (long)stride * 16) * EDIM, 0, 1);

    build_edge_A(A, hbf, eabf, tbf, sSrc[w], sDst[w], e0, lane);

    // diff_norm; col 304 unused by Wx (zero weight rows)
    if (lane < 16) {
      int s = sSrc[w][lane], d = sDst[w][lane];
      float dx = x[s * 3 + 0] - x[d * 3 + 0];
      float dy = x[s * 3 + 1] - x[d * 3 + 1];
      float dz = x[s * 3 + 2] - x[d * 3 + 2];
      float ds = dx * dx + dy * dy + dz * dz;
      float inv = 1.f / (__fsqrt_rn(ds + 1e-8f) + 1.f);
      sDiff[w][lane][0] = dx * inv;
      sDiff[w][lane][1] = dy * inv;
      sDiff[w][lane][2] = dz * inv;
      A[lane * KPAD + 304] = f2bf(0.f);
    }

    // layer 1
    for (int nt = 0; nt < 8; ++nt) {
      v8f acc = {};
      for (int kc = 0; kc < 10; ++kc) {
        v16bf a = load_a_frag(A, lane, kc * 32, KPAD);
        v16bf b = load_b_frag(sW1 + (kc * 8 + nt) * 512, lane);
        acc = wmma_bf16(a, b, acc);
      }
      int col = nt * 16 + (lane & 15);
      int mBase = (lane >> 4) * 8;
      float bb = sB1[col];
#pragma unroll
      for (int j = 0; j < 8; ++j)
        Hd[(mBase + j) * HDIM + col] = f2bf(siluf(acc[j] + bb));
    }
    // layer 2 (SiLU output, f32 -> Mb)
    for (int nt = 0; nt < 8; ++nt) {
      v8f acc = {};
      for (int kc = 0; kc < 4; ++kc) {
        v16bf a = load_a_frag(Hd, lane, kc * 32, HDIM);
        v16bf b = load_b_frag(sW2 + (kc * 8 + nt) * 512, lane);
        acc = wmma_bf16(a, b, acc);
      }
      int col = nt * 16 + (lane & 15);
      int mBase = (lane >> 4) * 8;
      float bb = sB2[col];
#pragma unroll
      for (int j = 0; j < 8; ++j)
        Mb[(mBase + j) * HDIM + col] = siluf(acc[j] + bb);
    }
    // coord weight = tanh(row . w3)
    if (lane < 16) {
      float dot = 0.f;
      const float* row = Mb + lane * HDIM;
      for (int n = 0; n < HDIM; ++n) dot += row[n] * sW3[n];
      sCw[w][lane] = tanhf(dot);
    }
    // scatter coord shift (16 edges x 3)
    for (int i = lane; i < 48; i += 32) {
      int m = i / 3, c = i - m * 3;
      float v = sDiff[w][m][c] * sCw[w][m] * 2.5f;
      atomAddF(&coord_agg[(long)sDst[w][m] * 3 + c], v);
    }
  }
}

// ---------------- node kernel: phi_h + residuals + x_out ----------------
__global__ __launch_bounds__(NW * 32, 1)
void k_node(const float* __restrict__ h, const __bf16* __restrict__ hbf,
            const float* __restrict__ x,
            const float* __restrict__ msg_agg, const float* __restrict__ coord_agg,
            const __bf16* __restrict__ W1f, const __bf16* __restrict__ W2f,
            const float* __restrict__ b1, const float* __restrict__ b2,
            float* __restrict__ outH, float* __restrict__ outX) {
  __shared__ alignas(16) __bf16 sW1[256 * HDIM];   // 64 KB
  __shared__ alignas(16) __bf16 sW2[HDIM * HDIM];  // 32 KB
  __shared__ float sB1[HDIM], sB2[HDIM];
  __shared__ alignas(16) __bf16 sA[NW][16 * 256];  // 64 KB
  __shared__ alignas(16) __bf16 sHid[NW][16 * HDIM];

  const int tid = threadIdx.x, lane = tid & 31, w = tid >> 5;
  copyW(sW1, W1f, 256 * HDIM, tid, NW * 32);
  copyW(sW2, W2f, HDIM * HDIM, tid, NW * 32);
  copyF(sB1, b1, HDIM, tid, NW * 32);
  copyF(sB2, b2, HDIM, tid, NW * 32);
  __syncthreads();

  __bf16* A  = &sA[w][0];
  __bf16* Hd = &sHid[w][0];

  const int nTiles = N_NODES / 16;
  const int gw = blockIdx.x * NW + w;
  const int stride = gridDim.x * NW;

  for (int t = gw; t < nTiles; t += stride) {
    const int n0 = t * 16;
    // A = [h | msg_agg], 16 rows x 256
    for (int i = lane; i < 256; i += 32) {
      int m = i >> 4, c = i & 15;
      cp16(hbf + (size_t)(n0 + m) * HDIM + c * 8, A + m * 256 + c * 8);
    }
    cp_fence();
    for (int i = lane; i < 512; i += 32) {
      int m = i >> 5, c = i & 31;
      float4 vm = ((const float4*)msg_agg)[(size_t)(n0 + m) * 32 + c];
      st4bf(A + m * 256 + HDIM + c * 4, vm);
    }
    // layer 1
    for (int nt = 0; nt < 8; ++nt) {
      v8f acc = {};
      for (int kc = 0; kc < 8; ++kc) {
        v16bf a = load_a_frag(A, lane, kc * 32, 256);
        v16bf b = load_b_frag(sW1 + (kc * 8 + nt) * 512, lane);
        acc = wmma_bf16(a, b, acc);
      }
      int col = nt * 16 + (lane & 15);
      int mBase = (lane >> 4) * 8;
      float bb = sB1[col];
#pragma unroll
      for (int j = 0; j < 8; ++j)
        Hd[(mBase + j) * HDIM + col] = f2bf(siluf(acc[j] + bb));
    }
    // layer 2 + residual -> outH
    for (int nt = 0; nt < 8; ++nt) {
      v8f acc = {};
      for (int kc = 0; kc < 4; ++kc) {
        v16bf a = load_a_frag(Hd, lane, kc * 32, HDIM);
        v16bf b = load_b_frag(sW2 + (kc * 8 + nt) * 512, lane);
        acc = wmma_bf16(a, b, acc);
      }
      int col = nt * 16 + (lane & 15);
      int mBase = (lane >> 4) * 8;
      float bb = sB2[col];
#pragma unroll
      for (int j = 0; j < 8; ++j) {
        long idx = (long)(n0 + mBase + j) * HDIM + col;
        outH[idx] = h[idx] + acc[j] + bb;
      }
    }
    // x_out
    if (lane < 16) {
      long n = n0 + lane;
      outX[n * 3 + 0] = x[n * 3 + 0] + coord_agg[n * 3 + 0];
      outX[n * 3 + 1] = x[n * 3 + 1] + coord_agg[n * 3 + 1];
      outX[n * 3 + 2] = x[n * 3 + 2] + coord_agg[n * 3 + 2];
    }
  }
}

extern "C" void kernel_launch(void* const* d_in, const int* in_sizes, int n_in,
                              void* d_out, int out_size, void* d_ws, size_t ws_size,
                              hipStream_t stream) {
  const float*     h      = (const float*)d_in[0];
  const float*     x      = (const float*)d_in[1];
  const long long* eidx   = (const long long*)d_in[2];
  const float*     eattr  = (const float*)d_in[3];
  const float*     temb   = (const float*)d_in[4];
  const float*     We_w1  = (const float*)d_in[5];
  const float*     We_b1  = (const float*)d_in[6];
  const float*     We_w2  = (const float*)d_in[7];
  const float*     We_b2  = (const float*)d_in[8];
  const float*     Watt_w = (const float*)d_in[9];
  const float*     Watt_b = (const float*)d_in[10];
  const float*     Wx_w1  = (const float*)d_in[11];
  const float*     Wx_b1  = (const float*)d_in[12];
  const float*     Wx_w2  = (const float*)d_in[13];
  const float*     Wx_b2  = (const float*)d_in[14];
  const float*     Wx_w3  = (const float*)d_in[15];
  const float*     Wh_w1  = (const float*)d_in[16];
  const float*     Wh_b1  = (const float*)d_in[17];
  const float*     Wh_w2  = (const float*)d_in[18];
  const float*     Wh_b2  = (const float*)d_in[19];
  (void)in_sizes; (void)n_in; (void)out_size; (void)ws_size;

  // workspace layout
  float*  msg_agg   = (float*)d_ws;
  float*  coord_agg = msg_agg + (size_t)N_NODES * HDIM;
  __bf16* frag      = (__bf16*)(coord_agg + (size_t)N_NODES * 3);
  __bf16* We1f = frag;
  __bf16* Wx1f = frag + 40960;
  __bf16* We2f = frag + 81920;
  __bf16* Wx2f = frag + 98304;
  __bf16* Wh1f = frag + 114688;
  __bf16* Wh2f = frag + 147456;
  __bf16* hbf  = frag + 163840;                       // N*H bf16
  __bf16* tbf  = hbf  + (size_t)N_NODES * HDIM;       // N*TD
  __bf16* eabf = tbf  + (size_t)N_NODES * TDIM;       // E*ED

  float* outH = (float*)d_out;
  float* outX = outH + (size_t)N_NODES * HDIM;

  int zn = N_NODES * HDIM + N_NODES * 3;
  k_zero<<<(zn + 255) / 256, 256, 0, stream>>>(msg_agg, zn);

  k_cvt<<<(N_NODES * HDIM + 255) / 256, 256, 0, stream>>>(h, hbf, N_NODES * HDIM);
  k_cvt<<<(N_NODES * TDIM + 255) / 256, 256, 0, stream>>>(temb, tbf, N_NODES * TDIM);
  k_cvt<<<(N_EDGES * EDIM + 255) / 256, 256, 0, stream>>>(eattr, eabf, N_EDGES * EDIM);

  k_repack<<<(KPAD * HDIM + 255) / 256, 256, 0, stream>>>(We_w1, We1f, KPAD, 305, 1);
  k_repack<<<(KPAD * HDIM + 255) / 256, 256, 0, stream>>>(Wx_w1, Wx1f, KPAD, 304, 2);
  k_repack<<<(HDIM * HDIM + 255) / 256, 256, 0, stream>>>(We_w2, We2f, HDIM, HDIM, 0);
  k_repack<<<(HDIM * HDIM + 255) / 256, 256, 0, stream>>>(Wx_w2, Wx2f, HDIM, HDIM, 0);
  k_repack<<<(256 * HDIM + 255) / 256, 256, 0, stream>>>(Wh_w1, Wh1f, 256, 256, 0);
  k_repack<<<(HDIM * HDIM + 255) / 256, 256, 0, stream>>>(Wh_w2, Wh2f, HDIM, HDIM, 0);

  k_edge_msg<<<512, NW * 32, 0, stream>>>(x, eidx, hbf, eabf, tbf,
                                          We1f, We2f, We_b1, We_b2,
                                          Watt_w, Watt_b, msg_agg);
  k_edge_coord<<<512, NW * 32, 0, stream>>>(x, eidx, hbf, eabf, tbf,
                                            Wx1f, Wx2f, Wx_b1, Wx_b2,
                                            Wx_w3, coord_agg);
  k_node<<<256, NW * 32, 0, stream>>>(h, hbf, x, msg_agg, coord_agg,
                                      Wh1f, Wh2f, Wh_b1, Wh_b2, outH, outX);
}
